// CoAtt_P_44083544326830
// MI455X (gfx1250) — matrix operand order repaired and checked by reading, not compiled, not run on hardware
//
#include <hip/hip_runtime.h>
#include <hip/hip_bf16.h>

// B=64, LQ=1024, LV=1024, DIM=256
// bf16 WMMA for both GEMMs; the [B,LQ,LV] tensor is never materialized
// (tanh is monotone => maxes commute with tanh). k_smax double-buffers the
// v tile via GLOBAL_LOAD_ASYNC_TO_LDS_B128 (ASYNCcnt) and keeps the wave's
// A-fragments resident in VGPRs across the whole LV loop.

typedef __attribute__((ext_vector_type(16))) __bf16 v16bf;
typedef __attribute__((ext_vector_type(8)))  __bf16 v8bf;
typedef __attribute__((ext_vector_type(8)))  float  v8f;
typedef __attribute__((ext_vector_type(4)))  int    v4i;

#define NB   64
#define LQ   1024
#define LV   1024
#define DIM  256
#define SP   264          // padded LDS row stride (bf16 elems) -> bank-conflict free
#define NEG_INF (-3.0e38f)

#define HAS_ASYNC_LDS __has_builtin(__builtin_amdgcn_global_load_async_to_lds_b128)

// ---------- async 16B global->LDS copy (falls back to sync copy) ----------
__device__ __forceinline__ void cp16_async(const __bf16* g, __bf16* l) {
#if HAS_ASYNC_LDS
  __builtin_amdgcn_global_load_async_to_lds_b128(
      (__attribute__((address_space(1))) v4i*)g,
      (__attribute__((address_space(3))) v4i*)l, 0, 0);
#else
  *(v8bf*)l = *(const v8bf*)g;
#endif
}
__device__ __forceinline__ void async_wait_all() {
#if HAS_ASYNC_LDS
  asm volatile("s_wait_asynccnt 0x0" ::: "memory");
#endif
}

// ---------- fragment helpers ----------
// A-fragment (16x32 bf16) from f32 row: chunks at +0..7 and +16..23 relative
// to (row + k0 + sel), sel = lane<16 ? 0 : 8.
__device__ __forceinline__ v16bf fragA_from_f32(const float* __restrict__ p) {
  float4 f0 = *(const float4*)(p);
  float4 f1 = *(const float4*)(p + 4);
  float4 f2 = *(const float4*)(p + 16);
  float4 f3 = *(const float4*)(p + 20);
  v16bf a;
  a[0]=(__bf16)f0.x;  a[1]=(__bf16)f0.y;  a[2]=(__bf16)f0.z;  a[3]=(__bf16)f0.w;
  a[4]=(__bf16)f1.x;  a[5]=(__bf16)f1.y;  a[6]=(__bf16)f1.z;  a[7]=(__bf16)f1.w;
  a[8]=(__bf16)f2.x;  a[9]=(__bf16)f2.y;  a[10]=(__bf16)f2.z; a[11]=(__bf16)f2.w;
  a[12]=(__bf16)f3.x; a[13]=(__bf16)f3.y; a[14]=(__bf16)f3.z; a[15]=(__bf16)f3.w;
  return a;
}

// A-fragment from LDS bf16: two 16B chunks at +0 and +16 elements
__device__ __forceinline__ v16bf fragA_lds(const __bf16* p) {
  v8bf lo = *(const v8bf*)(p);
  v8bf hi = *(const v8bf*)(p + 16);
  v16bf a;
#pragma unroll
  for (int i = 0; i < 8; ++i) { a[i] = lo[i]; a[8 + i] = hi[i]; }
  return a;
}

// B-fragment from LDS bf16: two 16B chunks at +0 and +16 elements
__device__ __forceinline__ v16bf fragB_lds(const __bf16* p) {
  v8bf lo = *(const v8bf*)(p);
  v8bf hi = *(const v8bf*)(p + 16);
  v16bf a;
#pragma unroll
  for (int i = 0; i < 8; ++i) { a[i] = lo[i]; a[8 + i] = hi[i]; }
  return a;
}

__device__ __forceinline__ v8f wmma_bf16(v16bf a, v16bf b, v8f c) {
  return __builtin_amdgcn_wmma_f32_16x16x32_bf16(false, a, false, b, (short)0, c,
                                                 false, false);
}

// ---------- conversion / init kernels ----------
__global__ __launch_bounds__(256) void k_cvt_v(const float* __restrict__ src,
                                               __bf16* __restrict__ dst) {
  size_t i = ((size_t)blockIdx.x * 256 + threadIdx.x) * 8;
  float4 f0 = *(const float4*)(src + i);
  float4 f1 = *(const float4*)(src + i + 4);
  v8bf o;
  o[0]=(__bf16)f0.x; o[1]=(__bf16)f0.y; o[2]=(__bf16)f0.z; o[3]=(__bf16)f0.w;
  o[4]=(__bf16)f1.x; o[5]=(__bf16)f1.y; o[6]=(__bf16)f1.z; o[7]=(__bf16)f1.w;
  *(v8bf*)(dst + i) = o;
}

__global__ __launch_bounds__(256) void k_cvt_wT(const float* __restrict__ w,
                                                __bf16* __restrict__ wT) {
  int e = blockIdx.x;              // 0..255
  int d = threadIdx.x;             // 0..255
  wT[(size_t)e * DIM + d] = (__bf16)w[(size_t)d * DIM + e];
}

__global__ __launch_bounds__(256) void k_init_max(float* __restrict__ p) {
  p[(size_t)blockIdx.x * 256 + threadIdx.x] = NEG_INF;
}

// ---------- kernel 1: qw[b] = q[b] (1024x256) @ w (256x256), bf16 out ----------
// One 16x16 output tile per wave; 8 waves / block; K=256 in 8 WMMA steps.
__global__ __launch_bounds__(256) void k_qw(const float* __restrict__ q,
                                            const __bf16* __restrict__ wT,
                                            __bf16* __restrict__ qw) {
  const int lane = threadIdx.x & 31;
  const int wv   = threadIdx.x >> 5;
  int tid = blockIdx.x * 8 + wv;       // 0..65535 tiles
  const int e_t = tid & 15;  tid >>= 4;   // DIM/16 = 16
  const int i_t = tid & 63;  tid >>= 6;   // LQ/16  = 64
  const int b   = tid;                    // 0..63
  const int lm   = lane & 15;
  const int selA = (lane < 16) ? 0 : 8;
  const int selB = (lane < 16) ? 0 : 8;

  const float*  arow = q  + ((size_t)b * LQ + i_t * 16 + lm) * DIM;
  const __bf16* bcol = wT + (size_t)(e_t * 16 + lm) * DIM;

  v8f c = {};
#pragma unroll
  for (int k0 = 0; k0 < DIM; k0 += 32) {
    v16bf a  = fragA_from_f32(arow + k0 + selA);
    v16bf bm = fragB_lds(bcol + k0 + selB);   // 16B aligned, chunks at +0 and +16
    c = wmma_bf16(a, bm, c);
  }
  // C layout: VGPR r -> row (r | lane>=16 ? +8), col = lane&15
  __bf16* orow = qw + ((size_t)b * LQ + i_t * 16) * DIM + e_t * 16 + lm;
  const int mrow = (lane < 16) ? 0 : 8;
#pragma unroll
  for (int r = 0; r < 8; ++r)
    orow[(size_t)(mrow + r) * DIM] = (__bf16)c[r];
}

// ---------- kernel 2: S = qw @ v^T, fused row/col max (S never stored) ------
// grid (LQ/64, B). Block owns 64 q-rows; loops all LV in 64-row v tiles,
// double-buffered via async global->LDS. A-fragments stay in VGPRs.
__global__ __launch_bounds__(256) void k_smax(const __bf16* __restrict__ qw,
                                              const __bf16* __restrict__ vb,
                                              float* __restrict__ m_q,
                                              float* __restrict__ m_v) {
  __shared__ __bf16 qs[64 * SP];
  __shared__ __bf16 vs[2][64 * SP];
  __shared__ float  smv[LV];
  __shared__ float  smq[64];

  const int t  = threadIdx.x;
  const int b  = blockIdx.y;
  const int q0 = blockIdx.x * 64;

  for (int i = t; i < LV; i += 256) smv[i] = NEG_INF;
  if (t < 64) smq[t] = NEG_INF;

  // issue async staging: qw tile + first v tile (each 2048 x 16B chunks)
  {
    const __bf16* srcq = qw + ((size_t)b * LQ + q0) * DIM;
    const __bf16* srcv = vb + ((size_t)b * LV) * DIM;
#pragma unroll
    for (int it = 0; it < 8; ++it) {
      int idx = t + it * 256;
      int row = idx >> 5, col = (idx & 31) << 3;
      cp16_async(srcq + (size_t)row * DIM + col, qs    + row * SP + col);
      cp16_async(srcv + (size_t)row * DIM + col, vs[0] + row * SP + col);
    }
  }
  async_wait_all();
  __syncthreads();

  const int lane = t & 31;
  const int wv   = t >> 5;
  const int r    = wv & 3;               // q-row tile within block (4)
  const int cp   = (wv >> 2) << 1;       // v-col tile pair base: 0 or 2
  const int lm   = lane & 15;
  const int selA = (lane < 16) ? 0 : 8;
  const int selB = (lane < 16) ? 0 : 8;

  // hoist A-fragments: invariant across the whole LV loop (8 x v16bf = 64 VGPRs)
  v16bf afr[8];
  {
    const __bf16* arow = qs + (r * 16 + lm) * SP + selA;
#pragma unroll
    for (int kk = 0; kk < 8; ++kk) afr[kk] = fragA_lds(arow + kk * 32);
  }

  float rmax[8];
#pragma unroll
  for (int i = 0; i < 8; ++i) rmax[i] = NEG_INF;

  for (int it2 = 0; it2 < 16; ++it2) {
    const int lv0 = it2 << 6;
    const __bf16* vcur = vs[it2 & 1];

    // prefetch next v tile into the other buffer (its readers finished at the
    // barrier that ended iteration it2-1)
    if (it2 < 15) {
      const __bf16* src = vb + ((size_t)b * LV + lv0 + 64) * DIM;
      __bf16* dst = vs[(it2 & 1) ^ 1];
#pragma unroll
      for (int it = 0; it < 8; ++it) {
        int idx = t + it * 256;
        int row = idx >> 5, col = (idx & 31) << 3;
        cp16_async(src + (size_t)row * DIM + col, dst + row * SP + col);
      }
    }

    v8f c0 = {}; v8f c1 = {};
    const __bf16* brow0 = vcur + (cp * 16 + lm) * SP + selB;
    const __bf16* brow1 = vcur + ((cp + 1) * 16 + lm) * SP + selB;
#pragma unroll
    for (int kk = 0; kk < 8; ++kk) {
      v16bf b0 = fragB_lds(brow0 + kk * 32);
      v16bf b1 = fragB_lds(brow1 + kk * 32);
      c0 = wmma_bf16(afr[kk], b0, c0);
      c1 = wmma_bf16(afr[kk], b1, c1);
    }

    // running row-max (per lane; cross-lane reduce deferred to the end)
    float cm0 = NEG_INF, cm1 = NEG_INF;
#pragma unroll
    for (int i = 0; i < 8; ++i) {
      rmax[i] = fmaxf(rmax[i], fmaxf(c0[i], c1[i]));
      cm0 = fmaxf(cm0, c0[i]);
      cm1 = fmaxf(cm1, c1[i]);
    }
    // column max: combine the two 8-row lane halves, then LDS atomic-max
    cm0 = fmaxf(cm0, __shfl_xor(cm0, 16, 32));
    cm1 = fmaxf(cm1, __shfl_xor(cm1, 16, 32));
    if (lane < 16) {
      atomicMax(&smv[lv0 + cp * 16 + lane], cm0);
      atomicMax(&smv[lv0 + (cp + 1) * 16 + lane], cm1);
    }

    async_wait_all();   // next buffer fully landed in LDS
    __syncthreads();    // all waves done reading vcur / done their atomics
  }

  // finish row max: reduce over the 16 columns held across each lane half
#pragma unroll
  for (int i = 0; i < 8; ++i) {
    float x = rmax[i];
    x = fmaxf(x, __shfl_xor(x, 1, 32));
    x = fmaxf(x, __shfl_xor(x, 2, 32));
    x = fmaxf(x, __shfl_xor(x, 4, 32));
    x = fmaxf(x, __shfl_xor(x, 8, 32));
    rmax[i] = x;
  }
  if (lm == 0) {  // lanes 0 (rows r*16+0..7) and 16 (rows r*16+8..15)
    const int mbase = r * 16 + ((lane < 16) ? 0 : 8);
#pragma unroll
    for (int i = 0; i < 8; ++i) atomicMax(&smq[mbase + i], rmax[i]);
  }
  __syncthreads();

  if (t < 64) m_q[(size_t)b * LQ + q0 + t] = smq[t];          // rows exclusive
  for (int i = t; i < LV; i += 256)
    atomicMax(&m_v[(size_t)b * LV + i], smv[i]);              // cols shared
}

// ---------- kernel 3: tanh -> softmax -> weighted sum -------------------
// grid (B, 2): y==0 -> att_q from (m_q, q); y==1 -> att_v from (m_v, v)
__global__ __launch_bounds__(256) void k_att(const float* __restrict__ m_q,
                                             const float* __restrict__ m_v,
                                             const float* __restrict__ q,
                                             const float* __restrict__ v,
                                             float* __restrict__ out) {
  __shared__ float p[1024];
  __shared__ float red[256];
  const int t = threadIdx.x;
  const int b = blockIdx.x;
  const int which = blockIdx.y;
  const float* mx  = which ? m_v : m_q;
  const float* src = which ? v : q;

  float s = 0.f;
  for (int i = t; i < 1024; i += 256) {
    // tanh in (-1,1) => exp safely bounded, no max-subtract needed
    float e = __expf(tanhf(mx[(size_t)b * 1024 + i]));
    p[i] = e; s += e;
  }
  red[t] = s;
  __syncthreads();
  for (int off = 128; off > 0; off >>= 1) {
    if (t < off) red[t] += red[t + off];
    __syncthreads();
  }
  const float inv = 1.0f / red[0];

  const float* col = src + (size_t)b * 1024 * DIM + t;  // coalesced over t
  float acc = 0.f;
#pragma unroll 4
  for (int i = 0; i < 1024; ++i) acc += p[i] * col[(size_t)i * DIM];
  out[(size_t)which * (NB * DIM) + (size_t)b * DIM + t] = acc * inv;
}

// ---------- launch ----------
extern "C" void kernel_launch(void* const* d_in, const int* in_sizes, int n_in,
                              void* d_out, int out_size, void* d_ws, size_t ws_size,
                              hipStream_t stream) {
  const float* q = (const float*)d_in[0];
  const float* v = (const float*)d_in[1];
  const float* w = (const float*)d_in[2];
  float* out = (float*)d_out;

  char* ws = (char*)d_ws;
  __bf16* v_bf  = (__bf16*)(ws);                         // 32 MB
  __bf16* qw_bf = (__bf16*)(ws + (size_t)33554432);      // 32 MB
  __bf16* wT    = (__bf16*)(ws + (size_t)67108864);      // 128 KB
  float*  m_q   = (float*) (ws + (size_t)67239936);      // 256 KB
  float*  m_v   = m_q + (size_t)NB * LQ;                 // 256 KB (contiguous)

  k_cvt_v  <<<8192, 256, 0, stream>>>(v, v_bf);
  k_cvt_wT <<<256,  256, 0, stream>>>(w, wT);
  k_init_max<<<512, 256, 0, stream>>>(m_q);              // covers m_q and m_v
  k_qw     <<<8192, 256, 0, stream>>>(q, wT, qw_bf);
  k_smax   <<<dim3(LQ / 64, NB), 256, 0, stream>>>(qw_bf, v_bf, m_q, m_v);
  k_att    <<<dim3(NB, 2), 256, 0, stream>>>(m_q, m_v, q, v, out);
}
